// RNN_83425444758301
// MI455X (gfx1250) — matrix operand order repaired and checked by reading, not compiled, not run on hardware
//
#include <hip/hip_runtime.h>
#include <stdint.h>
#include <math.h>

// ---------------------------------------------------------------------------
// 2-layer tanh RNN, B=32, L=2048, H=512 on MI455X (gfx1250, wave32, WMMA).
//
// Phase 1 (parallel):  pack weights to bf16 WMMA B-fragment order (2MB),
//                      init h fragment buffers from h0,
//                      Xproj[t] = x_t @ W_i0 + b_i0 for all t (big WMMA pass,
//                      stored bf16 in C-fragment order).
// Phase 2 (serial):    persistent 16-WG kernel; per step:
//                        L0: h0 = tanh(Xproj[t] + h0@W_h0 + b_h0)   (split-K)
//                        L1: h1 = tanh(h0@W_i1 + h1@W_h1 + b_i1+b_h1)
//                      grid barrier (atomic, monotonic) between phases;
//                      h images staged to LDS via global_load_async_to_lds.
// ---------------------------------------------------------------------------

#define B_   32
#define L_   2048
#define H_   512
#define NWG  16

typedef __attribute__((ext_vector_type(16))) __bf16 v16bf;
typedef __attribute__((ext_vector_type(8)))  __bf16 v8bf;
typedef __attribute__((ext_vector_type(8)))  float  v8f;

// workspace layout (bytes)
#define W_ELEMS  (4u*512u*512u)                 // 4 matrices, bf16, frag order
#define HBUF_EL  16384u                          // one (layer,buf) h image: 2rt*16kf*32lane*16
#define OFF_BAR  ((size_t)0)
#define OFF_W    ((size_t)256)
#define OFF_H    (OFF_W + (size_t)W_ELEMS*2u)
#define OFF_XP   (OFF_H + (size_t)4u*HBUF_EL*2u)
#define XP_EL    ((size_t)L_*64u*32u*8u)         // per t: 64 tiles * 32 lanes * 8 vals
#define WS_NEED  (OFF_XP + XP_EL*2u)             // ~69 MB

static __device__ __forceinline__ v8f vzero8() {
  v8f z = {0.f,0.f,0.f,0.f,0.f,0.f,0.f,0.f}; return z;
}
static __device__ __forceinline__ size_t wfrag_idx(unsigned mat, unsigned ct,
                                                   unsigned kf, unsigned lane) {
  return ((size_t)(((mat*32u + ct)*16u + kf)*32u + lane))*16u;
}
static __device__ __forceinline__ unsigned hfrag_off(unsigned rt, unsigned kf,
                                                     unsigned lane) {
  return ((rt*16u + kf)*32u + lane)*16u;         // bf16 element offset in one image
}

// ---- K1: pack fp32 weights -> bf16 WMMA B-fragments --------------------------
// B-frag (32x16 KxN): lane = half*16 + n ; values e<8 -> K=half*8+e, else 16+half*8+(e-8)
__global__ void k_pack_w(const float* __restrict__ Wi, const float* __restrict__ Wh,
                         __bf16* __restrict__ wf) {
  unsigned gid = blockIdx.x*256u + threadIdx.x;          // one bf16 element each
  unsigned e    = gid & 15u;
  unsigned lane = (gid >> 4) & 31u;
  unsigned kf   = (gid >> 9) & 15u;
  unsigned ct   = (gid >> 13) & 31u;
  unsigned mat  = gid >> 18;                             // 0:Wi0 1:Wi1 2:Wh0 3:Wh1
  unsigned k = kf*32u + ((e < 8u) ? (lane >> 4)*8u + e : 16u + (lane >> 4)*8u + (e - 8u));
  unsigned n = ct*16u + (lane & 15u);
  float v = (mat < 2u) ? Wi[(size_t)mat*H_*H_ + (size_t)k*H_ + n]
                       : Wh[(size_t)(mat-2u)*H_*H_ + (size_t)k*H_ + n];
  wf[gid] = (__bf16)v;
}

// ---- K2: init h fragment images from h0 (A-fragment order) -------------------
// A-frag (16x32 MxK): lane = half*16+m ; values e<8 -> K=half*8+e, else 16+half*8+(e-8)
__global__ void k_init_h(const float* __restrict__ h0, __bf16* __restrict__ hb) {
  unsigned gid = blockIdx.x*256u + threadIdx.x;          // < 2*2*16*32*16 = 32768
  unsigned e    = gid & 15u;
  unsigned lane = (gid >> 4) & 31u;
  unsigned kf   = (gid >> 9) & 15u;
  unsigned rt   = (gid >> 13) & 1u;
  unsigned l    = gid >> 14;
  unsigned m = lane & 15u, hf = lane >> 4;
  unsigned K = kf*32u + ((e < 8u) ? hf*8u + e : 16u + hf*8u + (e - 8u));
  unsigned row = rt*16u + m;
  float v = h0[((size_t)l*B_ + row)*H_ + K];
  hb[(size_t)(l*2u)*HBUF_EL + (gid & 16383u)] = (__bf16)v;  // cur buffer = 0
}

// ---- K3: Xproj[t] = x_t @ W_i0 + b_i0 (all t in parallel) --------------------
// one wave handles (t, rt, 4 consecutive col-tiles); A built from fp32 x on the fly
__global__ void __launch_bounds__(256) k_xproj(const float* __restrict__ x,
                                               const float* __restrict__ b_i,
                                               const __bf16* __restrict__ wf,
                                               __bf16* __restrict__ xp) {
  unsigned wid = threadIdx.x >> 5, lane = threadIdx.x & 31u;
  unsigned job = blockIdx.x*8u + wid;                    // < 2048*16 = 32768
  unsigned t  = job >> 4;
  unsigned rt = (job >> 3) & 1u;
  unsigned cg = job & 7u;                                // col-tile group (4 tiles)
  unsigned m = lane & 15u, hf = lane >> 4;
  const float* xr = x + ((size_t)(rt*16u + m)*L_ + t)*H_;
  v8f acc0 = vzero8(), acc1 = vzero8(), acc2 = vzero8(), acc3 = vzero8();
#pragma unroll 4
  for (unsigned kf = 0; kf < 16u; ++kf) {
    const float* p0 = xr + kf*32u + hf*8u;
    const float4 f0 = *(const float4*)(p0);
    const float4 f1 = *(const float4*)(p0 + 4);
    const float4 f2 = *(const float4*)(p0 + 16);
    const float4 f3 = *(const float4*)(p0 + 20);
    v16bf a;
    a[0]=(__bf16)f0.x;  a[1]=(__bf16)f0.y;  a[2]=(__bf16)f0.z;  a[3]=(__bf16)f0.w;
    a[4]=(__bf16)f1.x;  a[5]=(__bf16)f1.y;  a[6]=(__bf16)f1.z;  a[7]=(__bf16)f1.w;
    a[8]=(__bf16)f2.x;  a[9]=(__bf16)f2.y;  a[10]=(__bf16)f2.z; a[11]=(__bf16)f2.w;
    a[12]=(__bf16)f3.x; a[13]=(__bf16)f3.y; a[14]=(__bf16)f3.z; a[15]=(__bf16)f3.w;
    v16bf b0 = *(const v16bf*)(wf + wfrag_idx(0u, cg*4u+0u, kf, lane));
    v16bf b1 = *(const v16bf*)(wf + wfrag_idx(0u, cg*4u+1u, kf, lane));
    v16bf b2 = *(const v16bf*)(wf + wfrag_idx(0u, cg*4u+2u, kf, lane));
    v16bf b3 = *(const v16bf*)(wf + wfrag_idx(0u, cg*4u+3u, kf, lane));
    acc0 = __builtin_amdgcn_wmma_f32_16x16x32_bf16(false, a, false, b0, (short)0, acc0, false, false);
    acc1 = __builtin_amdgcn_wmma_f32_16x16x32_bf16(false, a, false, b1, (short)0, acc1, false, false);
    acc2 = __builtin_amdgcn_wmma_f32_16x16x32_bf16(false, a, false, b2, (short)0, acc2, false, false);
    acc3 = __builtin_amdgcn_wmma_f32_16x16x32_bf16(false, a, false, b3, (short)0, acc3, false, false);
  }
  v8f accs[4] = {acc0, acc1, acc2, acc3};
#pragma unroll
  for (unsigned q = 0; q < 4u; ++q) {
    unsigned ct = cg*4u + q, tile = ct*2u + rt, col = ct*16u + m;
    float bias = b_i[col];                               // layer 0 input bias
    v8bf o;
#pragma unroll
    for (int r = 0; r < 8; ++r) o[r] = (__bf16)(accs[q][r] + bias);
    *(v8bf*)(xp + ((size_t)((size_t)t*64u + tile)*32u + lane)*8u) = o;
  }
}

// ---- persistent recurrence kernel --------------------------------------------
static __device__ __forceinline__ void stage32k_async(uint32_t lds_base,
                                                      const __bf16* src,
                                                      unsigned tid) {
  uint64_t base = (uint64_t)(uintptr_t)src;
#pragma unroll
  for (unsigned j = 0; j < 8u; ++j) {
    uint32_t off = (tid + j*256u)*16u;                   // 256 thr * 8 * 16B = 32KB
    asm volatile("global_load_async_to_lds_b128 %0, %1, %2"
                 :: "v"(lds_base + off), "v"(off), "s"(base) : "memory");
  }
}

static __device__ __forceinline__ void grid_sync(unsigned* bar, unsigned target) {
  __syncthreads();
  if (threadIdx.x == 0) {
    __hip_atomic_fetch_add(bar, 1u, __ATOMIC_RELEASE, __HIP_MEMORY_SCOPE_AGENT);
    while (__hip_atomic_load(bar, __ATOMIC_ACQUIRE, __HIP_MEMORY_SCOPE_AGENT) < target)
      __builtin_amdgcn_s_sleep(2);
  }
  __syncthreads();
  __threadfence();                                       // invalidate stale lines
}

__global__ void __launch_bounds__(256, 1)
k_recur(const __bf16* __restrict__ wf, const __bf16* __restrict__ xp,
        __bf16* __restrict__ hb, const float* __restrict__ b_i,
        const float* __restrict__ b_h, float* __restrict__ out,
        unsigned* __restrict__ bar) {
  __shared__ __align__(16) unsigned char lds_h[32768];   // staged h image
  __shared__ float lds_part[4][32][8];                   // partner partial sums

  const unsigned tid = threadIdx.x, wid = tid >> 5, lane = tid & 31u;
  const unsigned tileInWg = wid >> 1, path = wid & 1u;
  const unsigned gtile = (blockIdx.x*8u + wid) >> 1;     // 0..63
  const unsigned rt = gtile & 1u, ct = gtile >> 1;
  const unsigned m = lane & 15u, hf = lane >> 4;
  const unsigned col = ct*16u + m;
  // mapping of a produced column `col` into next-step A-fragment coordinates
  const unsigned kf2  = col >> 5, kl = col & 31u;
  const unsigned e_st = ((kl >> 4) & 1u)*8u + (kl & 7u);
  const unsigned ha_st = (kl >> 3) & 1u;
  const float bh0 = b_h[col];
  const float b1  = b_i[H_ + col] + b_h[H_ + col];
  const uint32_t lds_base = (uint32_t)(uintptr_t)&lds_h[0];

  unsigned target = 0;
#pragma unroll 1
  for (unsigned t = 0; t < L_; ++t) {
    const unsigned cur = t & 1u, nxt = cur ^ 1u;
    __bf16* h0c = hb + (size_t)(0u + cur)*HBUF_EL;
    __bf16* h0n = hb + (size_t)(0u + nxt)*HBUF_EL;
    __bf16* h1c = hb + (size_t)(2u + cur)*HBUF_EL;
    __bf16* h1n = hb + (size_t)(2u + nxt)*HBUF_EL;

    // ---------------- layer 0: acc = h0 @ W_h0, split-K across wave pair -----
    stage32k_async(lds_base, h0c, tid);
    asm volatile("s_wait_asynccnt 0x0" ::: "memory");
    __syncthreads();
    v8f acc = vzero8();
#pragma unroll
    for (unsigned kk = 0; kk < 8u; ++kk) {
      unsigned kf = path*8u + kk;
      v16bf a = *(const v16bf*)(lds_h + hfrag_off(rt, kf, lane)*2u);
      v16bf b = *(const v16bf*)(wf + wfrag_idx(2u, ct, kf, lane));   // W_h0
      acc = __builtin_amdgcn_wmma_f32_16x16x32_bf16(false, a, false, b, (short)0, acc, false, false);
    }
    __syncthreads();
    if (path) {
#pragma unroll
      for (int r = 0; r < 8; ++r) lds_part[tileInWg][lane][r] = acc[r];
    }
    __syncthreads();
    if (!path) {
      v8bf xpv = *(const v8bf*)(xp + ((size_t)((size_t)t*64u + gtile)*32u + lane)*8u);
#pragma unroll
      for (int r = 0; r < 8; ++r) {
        float v = tanhf(acc[r] + lds_part[tileInWg][lane][r] + (float)xpv[r] + bh0);
        unsigned m2 = (unsigned)r + hf*8u;               // local row 0..15
        h0n[((rt*16u + kf2)*32u + (ha_st*16u + m2))*16u + e_st] = (__bf16)v;
        if (t == L_ - 1u)
          out[(size_t)B_*L_*H_ + (size_t)(rt*16u + m2)*H_ + col] = v;  // h_final L0
      }
    }
    grid_sync(bar, target += NWG);                       // publish h0_new

    // ---------------- layer 1: path0 = h0_new@W_i1, path1 = h1@W_h1 ----------
    stage32k_async(lds_base, h0n, tid);
    asm volatile("s_wait_asynccnt 0x0" ::: "memory");
    __syncthreads();
    acc = vzero8();
    if (!path) {
#pragma unroll
      for (unsigned kf = 0; kf < 16u; ++kf) {
        v16bf a = *(const v16bf*)(lds_h + hfrag_off(rt, kf, lane)*2u);
        v16bf b = *(const v16bf*)(wf + wfrag_idx(1u, ct, kf, lane)); // W_i1
        acc = __builtin_amdgcn_wmma_f32_16x16x32_bf16(false, a, false, b, (short)0, acc, false, false);
      }
    } else {
#pragma unroll
      for (unsigned kf = 0; kf < 16u; ++kf) {
        v16bf a = *(const v16bf*)(h1c + hfrag_off(rt, kf, lane));    // L2-hot
        v16bf b = *(const v16bf*)(wf + wfrag_idx(3u, ct, kf, lane)); // W_h1
        acc = __builtin_amdgcn_wmma_f32_16x16x32_bf16(false, a, false, b, (short)0, acc, false, false);
      }
    }
    __syncthreads();
    if (path) {
#pragma unroll
      for (int r = 0; r < 8; ++r) lds_part[tileInWg][lane][r] = acc[r];
    }
    __syncthreads();
    if (!path) {
#pragma unroll
      for (int r = 0; r < 8; ++r) {
        float v = tanhf(acc[r] + lds_part[tileInWg][lane][r] + b1);
        unsigned m2 = (unsigned)r + hf*8u;
        unsigned row = rt*16u + m2;
        h1n[((rt*16u + kf2)*32u + (ha_st*16u + m2))*16u + e_st] = (__bf16)v;
        out[((size_t)row*L_ + t)*H_ + col] = v;          // outs[b, t, h]
        if (t == L_ - 1u)
          out[(size_t)B_*L_*H_ + (size_t)B_*H_ + (size_t)row*H_ + col] = v; // h_final L1
      }
    }
    grid_sync(bar, target += NWG);                       // publish h1_new
  }
}

// ---------------------------------------------------------------------------
extern "C" void kernel_launch(void* const* d_in, const int* in_sizes, int n_in,
                              void* d_out, int out_size, void* d_ws, size_t ws_size,
                              hipStream_t stream) {
  const float* x  = (const float*)d_in[0];
  const float* h0 = (const float*)d_in[1];
  const float* Wi = (const float*)d_in[2];
  const float* bi = (const float*)d_in[3];
  const float* Wh = (const float*)d_in[4];
  const float* bh = (const float*)d_in[5];
  float* out = (float*)d_out;
  unsigned char* ws = (unsigned char*)d_ws;
  if (ws_size < WS_NEED) return;                         // need ~69 MB scratch

  unsigned* bar = (unsigned*)(ws + OFF_BAR);
  __bf16* wf  = (__bf16*)(ws + OFF_W);
  __bf16* hb  = (__bf16*)(ws + OFF_H);
  __bf16* xpj = (__bf16*)(ws + OFF_XP);

  hipMemsetAsync(bar, 0, 256, stream);                   // reset grid barrier
  k_pack_w<<<W_ELEMS/256u, 256, 0, stream>>>(Wi, Wh, wf);
  k_init_h<<<32768u/256u, 256, 0, stream>>>(h0, hb);
  k_xproj<<<4096, 256, 0, stream>>>(x, bi, wf, xpj);
  k_recur<<<NWG, 256, 0, stream>>>(wf, xpj, hb, bi, bh, out, bar);
  (void)in_sizes; (void)n_in; (void)out_size;
}
